// LiquidNetwork_61314953118335
// MI455X (gfx1250) — compile-verified, matrix operand order checked by reading, not run on hardware
//
#include <hip/hip_runtime.h>
#include <hip/hip_bf16.h>
#include <stdint.h>

// Problem constants (match reference)
#define Bsz 256
#define Tsz 512
#define Isz 64
#define Hsz 512
#define Osz 64
#define DTc 0.1f
#define EPSc 1e-6f

typedef __attribute__((ext_vector_type(16))) __bf16 v16bf;
typedef __attribute__((ext_vector_type(8)))  float  v8f;

__device__ __forceinline__ unsigned short f2bf(float f) {
    unsigned u = __float_as_uint(f);
    u += 0x7FFFu + ((u >> 16) & 1u);   // round-to-nearest-even
    return (unsigned short)(u >> 16);
}

union FragU { uint4 u[2]; v16bf v; };

// A-fragment (16x32 bf16, row-major source, row stride = rs elements).
// ISA layout: lanes 0-15 -> M=lane, K = kt*32 + {0..7, 16..23}
//             lanes 16-31 -> M=lane-16, K = kt*32 + {8..15, 24..31}
__device__ __forceinline__ v16bf load_afrag(const unsigned short* base, int kt, int lane, int rs) {
    const int m = lane & 15;
    const int half = (lane >> 4) * 8;
    FragU f;
    f.u[0] = *(const uint4*)(base + m * rs + kt * 32 + half);
    f.u[1] = *(const uint4*)(base + m * rs + kt * 32 + 16 + half);
    return f.v;
}

// B-fragment from pre-packed weights: per (ntile,ktile) 32 lanes x 16 bf16 contiguous.
__device__ __forceinline__ v16bf load_bfrag(const unsigned short* Wp, int nt, int kt, int lane, int ktiles) {
    const uint4* p = (const uint4*)(Wp + (((size_t)(nt * ktiles + kt)) * 32 + lane) * 16);
    FragU f;
    f.u[0] = p[0];
    f.u[1] = p[1];
    return f.v;
}

__device__ __forceinline__ v8f wmma_bf16(v16bf a, v16bf b, v8f c) {
    return __builtin_amdgcn_wmma_f32_16x16x32_bf16(false, a, false, b, (short)0, c, false, false);
}

// Async global->LDS copy of one 32KB slab (16 rows x 512 f32), 512 threads:
// each thread issues 4x GLOBAL_LOAD_ASYNC_TO_LDS_B128 (ASYNCcnt-tracked, CDNA5 TDM-class path).
__device__ __forceinline__ void async_copy_slab(const float* __restrict__ gsrc, float* lbuf, int tid) {
    uint32_t lbase = (uint32_t)(uintptr_t)lbuf + (uint32_t)tid * 16u;
    const char* g  = (const char*)gsrc + (size_t)tid * 16;
#pragma unroll
    for (int i = 0; i < 4; i++) {
        uint32_t laddr = lbase + (uint32_t)i * 8192u;
        uint64_t gaddr = (uint64_t)(uintptr_t)(g + (size_t)i * 8192);
        asm volatile("global_load_async_to_lds_b128 %0, %1, off"
                     :: "v"(laddr), "v"(gaddr)
                     : "memory");
    }
}

// ---------------- elementwise f32 -> bf16 ----------------
__global__ void cvt_f32_bf16_kernel(const float* __restrict__ in, unsigned short* __restrict__ out, int n) {
    int i = blockIdx.x * 256 + threadIdx.x;
    if (i < n) out[i] = f2bf(in[i]);
}

// ---------------- pack weight (R x K row-major f32) into WMMA-B fragment order ----------------
// out[((nt*(K/32)+kt)*32 + lane)*16 + e] = bf16(W[row][col])
//   row = nt*16 + (lane&15);  col = kt*32 + (e>>3)*16 + (lane>>4)*8 + (e&7)
__global__ void pack_weight_kernel(const float* __restrict__ W, unsigned short* __restrict__ out, int R, int K) {
    int idx = blockIdx.x * 256 + threadIdx.x;
    if (idx >= R * K) return;
    int e    = idx & 15;
    int lane = (idx >> 4) & 31;
    int gt   = idx >> 9;              // nt*(K/32)+kt
    int ktiles = K >> 5;
    int kt = gt % ktiles;
    int nt = gt / ktiles;
    int row = nt * 16 + (lane & 15);
    int col = kt * 32 + ((e >> 3) * 16) + ((lane >> 4) * 8) + (e & 7);
    out[idx] = f2bf(W[row * K + col]);
}

// ---------------- xw = x @ W_in0^T + b_in0, output layout [t][b][h] f32 ----------------
// grid.x = (B*T)/16 mtiles, block = 256 (8 waves), wave w handles ntiles 4w..4w+3
__global__ __launch_bounds__(256) void xw_gemm_kernel(const unsigned short* __restrict__ xb,
                                                      const unsigned short* __restrict__ W0p,
                                                      const float* __restrict__ b0v,
                                                      float* __restrict__ xw) {
    const int mt   = blockIdx.x;
    const int lane = threadIdx.x & 31;
    const int w    = threadIdx.x >> 5;
    const int lm   = lane & 15;
    const int lh   = lane >> 4;

    v8f acc[4] = {};
#pragma unroll
    for (int kt = 0; kt < 2; kt++) {
        v16bf a = load_afrag(xb + (size_t)mt * 16 * Isz, kt, lane, Isz);
#pragma unroll
        for (int j = 0; j < 4; j++) {
            v16bf b = load_bfrag(W0p, w * 4 + j, kt, lane, 2);
            acc[j] = wmma_bf16(a, b, acc[j]);
        }
    }
#pragma unroll
    for (int j = 0; j < 4; j++) {
        const int n = (w * 4 + j) * 16 + lm;
        const float bias = b0v[n];
#pragma unroll
        for (int v = 0; v < 8; v++) {
            int r = mt * 16 + v + lh * 8;     // global row in (B*T), r = b*T + t
            int t = r & (Tsz - 1);
            int b = r >> 9;
            xw[((size_t)t * Bsz + b) * Hsz + n] = acc[j][v] + bias;
        }
    }
}

// ---------------- serial scan over T ----------------
// grid = B/16 blocks, block = 512 threads (16 waves); wave w owns columns [32w, 32w+32)
__global__ __launch_bounds__(512) void scan_kernel(const float* __restrict__ xw,       // [T][B][H]
                                                   const unsigned short* __restrict__ A0p,
                                                   const unsigned short* __restrict__ W1p,
                                                   const unsigned short* __restrict__ A1p,
                                                   const unsigned short* __restrict__ Wop,
                                                   const float* __restrict__ b_in1,
                                                   const float* __restrict__ tau0,
                                                   const float* __restrict__ tau1,
                                                   const float* __restrict__ b_out,
                                                   float* __restrict__ y) {            // [B][T][O]
    __shared__ unsigned short h0b[16 * Hsz];     // 16 KB  bf16 mirror of h0
    __shared__ unsigned short h1b[16 * Hsz];     // 16 KB  bf16 mirror of h1
    __shared__ float xwbuf[2][16 * Hsz];         // 64 KB  double-buffered drive slab

    const int tid  = threadIdx.x;
    const int lane = tid & 31;
    const int w    = tid >> 5;      // 0..15
    const int lm   = lane & 15;
    const int lh   = lane >> 4;
    const int b0   = blockIdx.x * 16;

    for (int i = tid; i < 16 * Hsz; i += 512) { h0b[i] = 0; h1b[i] = 0; }

    // per-lane constants (column n is fixed per lane per ntile)
    float it0[2], it1[2], bi1[2];
    int   nc[2];
#pragma unroll
    for (int j = 0; j < 2; j++) {
        int n = (2 * w + j) * 16 + lm;
        nc[j]  = n;
        it0[j] = DTc / (fabsf(tau0[n]) + EPSc);
        it1[j] = DTc / (fabsf(tau1[n]) + EPSc);
        bi1[j] = b_in1[n];
    }
    const float bo = (w < 4) ? b_out[w * 16 + lm] : 0.0f;

    // prologue: async-prefetch slab t=0 into buffer 0
    async_copy_slab(xw + ((size_t)0 * Bsz + b0) * Hsz, &xwbuf[0][0], tid);
    __syncthreads();   // also covers the LDS zero-init

    v8f h0r[2] = {};   // persistent f32 hidden state (exact Euler in registers)
    v8f h1r[2] = {};

    for (int t = 0; t < Tsz; t++) {
        // issue async prefetch of the next drive slab (overlaps Phase A WMMA chain)
        int tn = (t + 1 < Tsz) ? (t + 1) : t;           // dummy re-issue on last step keeps counts uniform
        async_copy_slab(xw + ((size_t)tn * Bsz + b0) * Hsz, &xwbuf[(t + 1) & 1][0], tid);

        // ---- Phase A: acc = h0 @ A0^T (reads h0b) ----
        v8f acc[2] = {};
#pragma unroll
        for (int kt = 0; kt < 16; kt++) {
            v16bf a = load_afrag(h0b, kt, lane, Hsz);
            acc[0] = wmma_bf16(a, load_bfrag(A0p, 2 * w,     kt, lane, 16), acc[0]);
            acc[1] = wmma_bf16(a, load_bfrag(A0p, 2 * w + 1, kt, lane, 16), acc[1]);
        }
        // older 4 async ops (slab t) complete in order; allow the 4 just-issued to remain in flight
        asm volatile("s_wait_asynccnt 4" ::: "memory");
        __syncthreads();   // all h0b reads done + slab t visible from every wave

        // ---- h0 Euler update in registers; publish bf16 copy ----
        const float* xwt = &xwbuf[t & 1][0];
#pragma unroll
        for (int j = 0; j < 2; j++) {
#pragma unroll
            for (int v = 0; v < 8; v++) {
                int m = v + lh * 8;
                float d  = acc[j][v] + xwt[m * Hsz + nc[j]];
                float th = tanhf(d);
                float h  = h0r[j][v];
                h += it0[j] * (th - h);
                h0r[j][v] = h;
                h0b[m * Hsz + nc[j]] = f2bf(h);
            }
        }
        __syncthreads();   // new h0b visible

        // ---- Phase B: drive = h0_new @ W_in1^T ; acc2 = h1 @ A1^T ----
        v8f drv[2] = {};
#pragma unroll
        for (int kt = 0; kt < 16; kt++) {
            v16bf a = load_afrag(h0b, kt, lane, Hsz);
            drv[0] = wmma_bf16(a, load_bfrag(W1p, 2 * w,     kt, lane, 16), drv[0]);
            drv[1] = wmma_bf16(a, load_bfrag(W1p, 2 * w + 1, kt, lane, 16), drv[1]);
        }
        v8f ac2[2] = {};
#pragma unroll
        for (int kt = 0; kt < 16; kt++) {
            v16bf a = load_afrag(h1b, kt, lane, Hsz);
            ac2[0] = wmma_bf16(a, load_bfrag(A1p, 2 * w,     kt, lane, 16), ac2[0]);
            ac2[1] = wmma_bf16(a, load_bfrag(A1p, 2 * w + 1, kt, lane, 16), ac2[1]);
        }
        __syncthreads();   // all h1b reads done

        // ---- h1 Euler update; publish bf16 copy ----
#pragma unroll
        for (int j = 0; j < 2; j++) {
#pragma unroll
            for (int v = 0; v < 8; v++) {
                int m = v + lh * 8;
                float s  = ac2[j][v] + drv[j][v] + bi1[j];
                float th = tanhf(s);
                float h  = h1r[j][v];
                h += it1[j] * (th - h);
                h1r[j][v] = h;
                h1b[m * Hsz + nc[j]] = f2bf(h);
            }
        }
        __syncthreads();   // new h1b visible

        // ---- Phase C: y = h1_new @ W_out^T + b_out (waves 0..3, O=64) ----
        if (w < 4) {
            v8f ya = {};
#pragma unroll
            for (int kt = 0; kt < 16; kt++) {
                v16bf a = load_afrag(h1b, kt, lane, Hsz);
                ya = wmma_bf16(a, load_bfrag(Wop, w, kt, lane, 16), ya);
            }
#pragma unroll
            for (int v = 0; v < 8; v++) {
                int m = v + lh * 8;
                y[((size_t)(b0 + m) * Tsz + t) * Osz + w * 16 + lm] = ya[v] + bo;
            }
        }
    }
}

extern "C" void kernel_launch(void* const* d_in, const int* in_sizes, int n_in,
                              void* d_out, int out_size, void* d_ws, size_t ws_size,
                              hipStream_t stream) {
    (void)in_sizes; (void)n_in; (void)out_size; (void)ws_size;

    const float* x_seq = (const float*)d_in[0];   // (B,T,I)
    const float* W_in0 = (const float*)d_in[1];   // (H,I)
    const float* b_in0 = (const float*)d_in[2];   // (H,)
    const float* A0    = (const float*)d_in[3];   // (H,H)
    const float* tau0  = (const float*)d_in[4];   // (H,)
    const float* W_in1 = (const float*)d_in[5];   // (H,H)
    const float* b_in1 = (const float*)d_in[6];   // (H,)
    const float* A1    = (const float*)d_in[7];   // (H,H)
    const float* tau1  = (const float*)d_in[8];   // (H,)
    const float* W_out = (const float*)d_in[9];   // (O,H)
    const float* b_out = (const float*)d_in[10];  // (O,)
    float* y = (float*)d_out;                     // (B,T,O)

    // workspace layout
    char* ws = (char*)d_ws;
    float*          xw  = (float*)ws;                                   // T*B*H f32
    size_t off = (size_t)Tsz * Bsz * Hsz * sizeof(float);
    unsigned short* xb  = (unsigned short*)(ws + off); off += (size_t)Bsz * Tsz * Isz * 2;
    unsigned short* W0p = (unsigned short*)(ws + off); off += (size_t)Hsz * Isz * 2;
    unsigned short* A0p = (unsigned short*)(ws + off); off += (size_t)Hsz * Hsz * 2;
    unsigned short* W1p = (unsigned short*)(ws + off); off += (size_t)Hsz * Hsz * 2;
    unsigned short* A1p = (unsigned short*)(ws + off); off += (size_t)Hsz * Hsz * 2;
    unsigned short* Wop = (unsigned short*)(ws + off); off += (size_t)Osz * Hsz * 2;

    // 1) x_seq -> bf16
    {
        int n = Bsz * Tsz * Isz;
        cvt_f32_bf16_kernel<<<(n + 255) / 256, 256, 0, stream>>>(x_seq, xb, n);
    }
    // 2) pack weights into WMMA-B fragment order (bf16)
    pack_weight_kernel<<<(Hsz * Isz + 255) / 256, 256, 0, stream>>>(W_in0, W0p, Hsz, Isz);
    pack_weight_kernel<<<(Hsz * Hsz + 255) / 256, 256, 0, stream>>>(A0,    A0p, Hsz, Hsz);
    pack_weight_kernel<<<(Hsz * Hsz + 255) / 256, 256, 0, stream>>>(W_in1, W1p, Hsz, Hsz);
    pack_weight_kernel<<<(Hsz * Hsz + 255) / 256, 256, 0, stream>>>(A1,    A1p, Hsz, Hsz);
    pack_weight_kernel<<<(Osz * Hsz + 255) / 256, 256, 0, stream>>>(W_out, Wop, Osz, Hsz);

    // 3) xw = x @ W_in0^T + b_in0   (layout [t][b][h])
    xw_gemm_kernel<<<(Bsz * Tsz) / 16, 256, 0, stream>>>(xb, W0p, b_in0, xw);

    // 4) serial liquid scan with WMMA per-step GEMMs + async drive prefetch
    scan_kernel<<<Bsz / 16, 512, 0, stream>>>(xw, A0p, W1p, A1p, Wop,
                                              b_in1, tau0, tau1, b_out, y);
}
